// CapsuleLayer_60601988546710
// MI455X (gfx1250) — matrix operand order repaired and checked by reading, not compiled, not run on hardware
//
#include <hip/hip_runtime.h>
#include <hip/hip_bf16.h>

// ---------------------------------------------------------------------------
// CapsuleLayer for MI455X (gfx1250, wave32).
//   B=64, L=512, D=1024, C=32, O=64, ROUTING_ITERS=3
//   Stage 1: u_hat = inputs(32768x1024) @ fc_w(1024x2048) + fc_b   [bf16 WMMA]
//   Stage 2: 3 routing iterations (softmax / s-reduce / squash / b-update)
// Workspace layout (bytes):
//   [0, 256MiB)          u_hat  fp32 (32768 x 2048)
//   [256MiB, +4MiB)      b_ij   fp32 (64 x 512 x 32)
//   [260MiB, +4MiB)      s_part fp32 (64 x 8 x 32 x 64)  (8 L-chunks per batch)
// ---------------------------------------------------------------------------

typedef __attribute__((ext_vector_type(16))) __bf16 v16bf;
typedef __attribute__((ext_vector_type(8)))  float  v8f;

#define M_TOT 32768
#define K_TOT 1024
#define N_TOT 2048
#define TM 128
#define TN 128
#define TK 32
#define LPAD 8   // ushort padding per LDS row (keeps 16B alignment, skews banks)

__device__ __forceinline__ unsigned int f2bf_bits(float f) {
    unsigned int u = __float_as_uint(f);
    return (u + 0x7FFFu + ((u >> 16) & 1u)) >> 16;   // round-to-nearest-even
}
__device__ __forceinline__ unsigned int pack2bf(float lo, float hi) {
    return f2bf_bits(lo) | (f2bf_bits(hi) << 16);
}

union FragA { uint4 q[2]; v16bf v; };

// ------------------------- GEMM: u_hat = A*W + bias -------------------------
__global__ __launch_bounds__(256) void gemm_uhat(
    const float* __restrict__ A,     // 32768 x 1024 (row-major, K contiguous)
    const float* __restrict__ W,     // 1024 x 2048  (row-major, N contiguous)
    const float* __restrict__ bias,  // 2048
    float* __restrict__ C)           // 32768 x 2048
{
    __shared__ unsigned short As[TM][TK + LPAD];  // [m][k]  bf16 bits
    __shared__ unsigned short Bs[TN][TK + LPAD];  // [n][k]  (transposed) bf16 bits

    const int tid  = threadIdx.x;
    const int wave = tid >> 5;
    const int lane = tid & 31;
    const int m0   = blockIdx.y * TM;   // gridDim.y = 256
    const int n0   = blockIdx.x * TN;   // gridDim.x = 16

    // wave tile: 32 (m) x 64 (n); waves arranged 4 (m) x 2 (n)
    const int wm = (wave >> 1) * 32;
    const int wn = (wave & 1) * 64;

    v8f acc[2][4];
    #pragma unroll
    for (int i = 0; i < 2; ++i)
        #pragma unroll
        for (int j = 0; j < 4; ++j)
            #pragma unroll
            for (int r = 0; r < 8; ++r) acc[i][j][r] = 0.0f;

    const int half = lane >> 4;    // 0: lanes 0-15, 1: lanes 16-31
    const int l16  = lane & 15;

    for (int k0 = 0; k0 < K_TOT; k0 += TK) {
        __syncthreads();
        // --- A tile 128x32 fp32 -> packed bf16x2; float2 global, b32 DS ---
        // 2048 k-pairs: m = idx>>4, kp = (idx&15)*2  (adjacent in k)
        #pragma unroll
        for (int i = 0; i < 8; ++i) {
            int idx = tid + 256 * i;            // 0..2047
            int m = idx >> 4, kp = (idx & 15) << 1;
            const float2 a2 = *(const float2*)(A + (size_t)(m0 + m) * K_TOT + k0 + kp);
            *(unsigned int*)&As[m][kp] = pack2bf(a2.x, a2.y);
        }
        // --- W tile 32x128 transposed into Bs[n][k], packed along k ---
        // 2048 k-pairs: n = idx&127, kp = (idx>>7)*2 (two coalesced row reads)
        #pragma unroll
        for (int i = 0; i < 8; ++i) {
            int idx = tid + 256 * i;            // 0..2047
            int n = idx & 127, kp = (idx >> 7) << 1;
            const float* wp = W + (size_t)(k0 + kp) * N_TOT + n0 + n;
            *(unsigned int*)&Bs[n][kp] = pack2bf(wp[0], wp[N_TOT]);
        }
        __syncthreads();

        // --- fragments (documented CDNA5 16-bit A 16x32 / B 32x16 layouts) ---
        // A lane(0-15): M=lane, K = {half*8 + 0..7, 16 + half*8 + 0..7}
        v16bf afrag[2];
        #pragma unroll
        for (int i = 0; i < 2; ++i) {
            const unsigned short* ap = &As[wm + i * 16 + l16][0];
            FragA fa;
            fa.q[0] = *(const uint4*)(ap + half * 8);
            fa.q[1] = *(const uint4*)(ap + 16 + half * 8);
            afrag[i] = fa.v;
        }
        // B lane: col = lane&15, K = {half*8 + 0..7, 16 + half*8 + 0..7}
        v16bf bfrag[4];
        #pragma unroll
        for (int j = 0; j < 4; ++j) {
            const unsigned short* bp = &Bs[wn + j * 16 + l16][0];
            FragA fb;
            fb.q[0] = *(const uint4*)(bp + half * 8);
            fb.q[1] = *(const uint4*)(bp + 16 + half * 8);
            bfrag[j] = fb.v;
        }
        #pragma unroll
        for (int i = 0; i < 2; ++i)
            #pragma unroll
            for (int j = 0; j < 4; ++j)
                acc[i][j] = __builtin_amdgcn_wmma_f32_16x16x32_bf16(
                    false, afrag[i], false, bfrag[j],
                    (short)0, acc[i][j], false, false);
    }

    // --- epilogue: C/D layout VGPR r -> (M = half*8 + r), N = lane&15 ---
    #pragma unroll
    for (int j = 0; j < 4; ++j) {
        int n = n0 + wn + j * 16 + l16;
        float bv = bias[n];
        #pragma unroll
        for (int i = 0; i < 2; ++i) {
            int mbase = m0 + wm + i * 16 + half * 8;
            #pragma unroll
            for (int r = 0; r < 8; ++r)
                C[(size_t)(mbase + r) * N_TOT + n] = acc[i][j][r] + bv;
        }
    }
}

// ------------------------------- zero fill ---------------------------------
__global__ void zero_f(float* __restrict__ p, int n) {
    int i = blockIdx.x * blockDim.x + threadIdx.x;
    if (i < n) p[i] = 0.0f;
}

// ------------- s-pass: softmax(b) then s_part[b][chunk] = sum_l c*u_hat -----
__global__ __launch_bounds__(256) void s_pass(
    const float* __restrict__ uhat,   // (64,512,32,64)
    const float* __restrict__ bij,    // (64,512,32)
    float* __restrict__ s_part)       // (64,8,32,64)
{
    __shared__ float c_lds[64][32];
    const int t   = threadIdx.x;
    const int b   = blockIdx.x >> 3;          // 512 blocks: (batch, chunk)
    const int ch  = blockIdx.x & 7;
    const int l0  = ch * 64;

    if (t < 64) {
        const float* brow = bij + ((size_t)(b * 512 + l0 + t)) * 32;
        float r[32];
        float mx = -1e30f;
        #pragma unroll
        for (int c = 0; c < 32; ++c) { r[c] = brow[c]; mx = fmaxf(mx, r[c]); }
        float ssum = 0.0f;
        #pragma unroll
        for (int c = 0; c < 32; ++c) { r[c] = __expf(r[c] - mx); ssum += r[c]; }
        float inv = 1.0f / ssum;
        #pragma unroll
        for (int c = 0; c < 32; ++c) c_lds[t][c] = r[c] * inv;
    }
    __syncthreads();

    float acc[8];
    #pragma unroll
    for (int j = 0; j < 8; ++j) acc[j] = 0.0f;

    const float* up = uhat + ((size_t)(b * 512 + l0)) * 2048;
    for (int l = 0; l < 64; ++l) {
        const float* row = up + (size_t)l * 2048;   // fully coalesced 8KB/row
        #pragma unroll
        for (int j = 0; j < 8; ++j) {
            int idx = t + 256 * j;                  // 0..2047 -> (c = idx>>6, o = idx&63)
            acc[j] += c_lds[l][idx >> 6] * row[idx];
        }
    }
    float* sp = s_part + ((size_t)(b * 8 + ch)) * 2048;
    #pragma unroll
    for (int j = 0; j < 8; ++j) sp[t + 256 * j] = acc[j];
}

// --------------- squash: v = squash(sum_ch s_part) -> d_out ----------------
__global__ void squash_reduce(const float* __restrict__ s_part,
                              float* __restrict__ v)     // (64,32,64)
{
    int gid  = blockIdx.x * blockDim.x + threadIdx.x;
    int w    = gid >> 5;            // (b*32 + c), 2048 waves total
    int lane = threadIdx.x & 31;
    if (w >= 2048) return;
    int b = w >> 5, c = w & 31;

    float x0 = 0.0f, x1 = 0.0f;
    #pragma unroll
    for (int ch = 0; ch < 8; ++ch) {
        const float* p = s_part + ((size_t)(b * 8 + ch)) * 2048 + c * 64;
        x0 += p[lane];
        x1 += p[lane + 32];
    }
    float sq = x0 * x0 + x1 * x1;
    #pragma unroll
    for (int off = 16; off > 0; off >>= 1) sq += __shfl_xor(sq, off, 32);
    float scale = (sq / (1.0f + sq)) * rsqrtf(sq + 1e-8f);
    v[(size_t)w * 64 + lane]      = scale * x0;
    v[(size_t)w * 64 + lane + 32] = scale * x1;
}

// --------------- b-update: b_ij += sum_o u_hat * v -------------------------
__global__ void b_update(const float* __restrict__ uhat,
                         const float* __restrict__ v,
                         float* __restrict__ bij)
{
    size_t i = (size_t)blockIdx.x * 256 + threadIdx.x;   // 0 .. B*L*C-1
    int c = (int)(i & 31);
    size_t bl = i >> 5;
    int b = (int)(bl >> 9);
    const float4* u  = (const float4*)(uhat + (bl * 32 + c) * 64);
    const float4* vv = (const float4*)(v + ((size_t)(b * 32 + c)) * 64);
    float dot = 0.0f;
    #pragma unroll
    for (int j = 0; j < 16; ++j) {
        float4 a = u[j], w = vv[j];
        dot += a.x * w.x + a.y * w.y + a.z * w.z + a.w * w.w;
    }
    bij[i] += dot;
}

// ---------------------------------------------------------------------------
extern "C" void kernel_launch(void* const* d_in, const int* in_sizes, int n_in,
                              void* d_out, int out_size, void* d_ws, size_t ws_size,
                              hipStream_t stream) {
    const float* inputs = (const float*)d_in[0];   // (64,512,1024)
    const float* fc_w   = (const float*)d_in[1];   // (1024,2048)
    const float* fc_b   = (const float*)d_in[2];   // (2048,)
    float* out = (float*)d_out;                    // (64,32,64)

    char* ws = (char*)d_ws;                        // needs ~264 MiB
    float* uhat   = (float*)(ws);                                    // 256 MiB
    float* bij    = (float*)(ws + (size_t)268435456);                // 4 MiB
    float* s_part = (float*)(ws + (size_t)268435456 + 4194304);      // 4 MiB

    // Stage 1: GEMM with bf16 WMMA
    gemm_uhat<<<dim3(16, 256), 256, 0, stream>>>(inputs, fc_w, fc_b, uhat);

    // b_ij = 0
    zero_f<<<4096, 256, 0, stream>>>(bij, 64 * 512 * 32);

    // Stage 2: 3 routing iterations (skip dead b-update on last iter)
    for (int it = 0; it < 3; ++it) {
        s_pass<<<512, 256, 0, stream>>>(uhat, bij, s_part);
        squash_reduce<<<256, 256, 0, stream>>>(s_part, out);
        if (it < 2) b_update<<<4096, 256, 0, stream>>>(uhat, out, bij);
    }
}
